// Transformer2_21474836480547
// MI455X (gfx1250) — matrix operand order repaired and checked by reading, not compiled, not run on hardware
//
#include <hip/hip_runtime.h>
#include <cmath>
#include <cstddef>

// ---------------------------------------------------------------------------
// Transformer2 forward for MI455X (gfx1250, wave32, WMMA).
//
// - Generic GEMM: f32 in/out, f16 operands staged in LDS (A row-major, B
//   transposed so per-lane fragments are contiguous b128 LDS loads), f32
//   accumulation with V_WMMA_F32_16X16X32_F16. Block tile 128x64, 8 waves,
//   4 WMMAs per wave per K-step, software-pipelined global->LDS staging.
// - Fused FFN: relu(x@W1+b1)@W2+b2 with the hidden tensor kept entirely in
//   LDS (f16), killing the ~300MB/pass hidden-activation HBM round-trip.
// - Attention softmaxes are clipped to [-5,5] by the reference, so exp() is
//   applied directly (no max pass).
// ---------------------------------------------------------------------------

typedef __attribute__((ext_vector_type(4)))  _Float16 v4h;
typedef __attribute__((ext_vector_type(8)))  _Float16 v8h;
typedef __attribute__((ext_vector_type(16))) _Float16 v16h;
typedef __attribute__((ext_vector_type(4)))  float    v4f;
typedef __attribute__((ext_vector_type(8)))  float    v8f;

#define CAT16(lo, hp)                                                        \
    __builtin_shufflevector(lo, hp, 0, 1, 2, 3, 4, 5, 6, 7, 8, 9, 10, 11,   \
                            12, 13, 14, 15)
#define WMMA(a, b, c)                                                        \
    __builtin_amdgcn_wmma_f32_16x16x32_f16(false, a, false, b, (short)0, c, \
                                           false, false)

#define N_   2048
#define R_   128
#define T_   18
#define E_   128
#define H_   8
#define E2_  256
#define NEGV (-1000000.0f)

#define BM 128
#define BN 64
#define BK 32
#define LDP (BK + 8)   // LDS row pitch in halves (80B, 16B-aligned)

// ------------------------------ GEMM (WMMA) --------------------------------
// C[M x Nc] = act(A[M x K] @ B[K x Nc] + bias)
// act: 0 none, 1 relu, 2 sigmoid. bias may be null.
// Assumes K % 4 == 0 (true for every GEMM in this graph: 16..1024).
__global__ void __launch_bounds__(256)
gemm_f16wmma(const float* __restrict__ A, const float* __restrict__ B,
             const float* __restrict__ bias, float* __restrict__ C,
             int M, int Nc, int K, int act)
{
    __shared__ _Float16 As[BM][LDP];    // A tile, row-major (row x K)
    __shared__ _Float16 BsT[BN][LDP];   // B tile, transposed (col x K)

    const int bm   = blockIdx.x * BM;
    const int bn   = blockIdx.y * BN;
    const int tid  = threadIdx.x;
    const int wave = tid >> 5;
    const int lane = tid & 31;
    const int wm   = (wave & 3) * 32;   // wave sub-tile row
    const int wn   = (wave >> 2) * 32;  // wave sub-tile col
    const int l    = lane & 15;
    const int hi   = lane >> 4;

    v4f   a_reg[4];
    float b_reg[2][4];

    auto load_tile = [&](int k0) {
#pragma unroll
        for (int j = 0; j < 4; ++j) {
            int idx = j * 256 + tid;
            int r = idx >> 3, c = (idx & 7) * 4;
            int gr = bm + r, gc = k0 + c;
            if (gr < M && gc < K)
                a_reg[j] = *(const v4f*)(A + (size_t)gr * K + gc);
            else
                a_reg[j] = (v4f){0.f, 0.f, 0.f, 0.f};
        }
#pragma unroll
        for (int j = 0; j < 2; ++j) {
            int krb = (j * 4 + (tid >> 6)) * 4;   // 0,4,...,28
            int c   = tid & 63;
            int gk = k0 + krb, gc = bn + c;
#pragma unroll
            for (int i = 0; i < 4; ++i)
                b_reg[j][i] = (gc < Nc && (gk + i) < K)
                                  ? B[(size_t)(gk + i) * Nc + gc] : 0.f;
        }
    };
    auto store_tile = [&]() {
#pragma unroll
        for (int j = 0; j < 4; ++j) {
            int idx = j * 256 + tid;
            int r = idx >> 3, c = (idx & 7) * 4;
            v4h h = {(_Float16)a_reg[j][0], (_Float16)a_reg[j][1],
                     (_Float16)a_reg[j][2], (_Float16)a_reg[j][3]};
            *(v4h*)&As[r][c] = h;
        }
#pragma unroll
        for (int j = 0; j < 2; ++j) {
            int krb = (j * 4 + (tid >> 6)) * 4;
            int c   = tid & 63;
            v4h h = {(_Float16)b_reg[j][0], (_Float16)b_reg[j][1],
                     (_Float16)b_reg[j][2], (_Float16)b_reg[j][3]};
            *(v4h*)&BsT[c][krb] = h;
        }
    };

    v8f acc[2][2];
#pragma unroll
    for (int i = 0; i < 2; ++i)
#pragma unroll
        for (int j = 0; j < 2; ++j)
            acc[i][j] = (v8f){0.f, 0.f, 0.f, 0.f, 0.f, 0.f, 0.f, 0.f};

    load_tile(0);
    for (int k0 = 0; k0 < K; k0 += BK) {
        store_tile();
        __syncthreads();
        if (k0 + BK < K) load_tile(k0 + BK);      // prefetch next tile

        const int ra0 = wm + l, ra1 = wm + 16 + l;
        const int cb0 = wn + l, cb1 = wn + 16 + l;
        v8h x0 = *(const v8h*)&As[ra0][hi * 8];
        v8h x1 = *(const v8h*)&As[ra0][16 + hi * 8];
        v8h y0 = *(const v8h*)&As[ra1][hi * 8];
        v8h y1 = *(const v8h*)&As[ra1][16 + hi * 8];
        v8h z0 = *(const v8h*)&BsT[cb0][hi * 8];
        v8h z1 = *(const v8h*)&BsT[cb0][16 + hi * 8];
        v8h w0 = *(const v8h*)&BsT[cb1][hi * 8];
        v8h w1 = *(const v8h*)&BsT[cb1][16 + hi * 8];
        v16h a0 = CAT16(x0, x1);
        v16h a1 = CAT16(y0, y1);
        v16h b0 = CAT16(z0, z1);
        v16h b1 = CAT16(w0, w1);

        acc[0][0] = WMMA(a0, b0, acc[0][0]);
        acc[0][1] = WMMA(a0, b1, acc[0][1]);
        acc[1][0] = WMMA(a1, b0, acc[1][0]);
        acc[1][1] = WMMA(a1, b1, acc[1][1]);
        __syncthreads();
    }

#pragma unroll
    for (int ti = 0; ti < 2; ++ti) {
#pragma unroll
        for (int tj = 0; tj < 2; ++tj) {
            int col = bn + wn + tj * 16 + l;
            if (col >= Nc) continue;
            const float bv = bias ? bias[col] : 0.f;
#pragma unroll
            for (int r = 0; r < 8; ++r) {
                int row = bm + wm + ti * 16 + hi * 8 + r;
                if (row < M) {
                    float v = acc[ti][tj][r] + bv;
                    if (act == 1)      v = fmaxf(v, 0.f);
                    else if (act == 2) v = 1.f / (1.f + expf(-v));
                    C[(size_t)row * Nc + col] = v;
                }
            }
        }
    }
}

// ------------------------------ fused FFN ----------------------------------
// Out[M x E] = relu(X[M x E] @ W1[E x 4E] + b1) @ W2[4E x E] + b2
// Requires M % 64 == 0 (all call sites). Hidden activations live in LDS only.
// Per block: 64-row tile. Per 32-wide hidden chunk:
//   stage1: 8 waves each produce one 16x16 h-tile (E/32 WMMAs, K = E)
//   stage2: 8 waves each accumulate a 16 x (E/2) output strip (E/32 WMMAs)
// W1T / W2T chunks time-share one LDS buffer (59KB total static LDS @E=256).
template <int E>
__global__ void __launch_bounds__(256)
ffn_fused(const float* __restrict__ X, const float* __restrict__ W1,
          const float* __restrict__ b1, const float* __restrict__ W2,
          const float* __restrict__ b2, float* __restrict__ Out)
{
    constexpr int XP = E + 8;                        // xs pitch (halves)
    constexpr int HP = 40;                           // hs / W2T pitch
    constexpr int WB = (32 * XP > E * HP) ? 32 * XP : E * HP;
    __shared__ _Float16 xs[64][XP];                  // x tile (A layout)
    __shared__ _Float16 hs[64][HP];                  // hidden chunk (A layout)
    __shared__ _Float16 wbuf[WB];                    // W1T chunk / W2T chunk

    const int bm   = blockIdx.x * 64;
    const int tid  = threadIdx.x;
    const int wave = tid >> 5;
    const int lane = tid & 31;
    const int l = lane & 15, hi = lane >> 4;
    const int rw = (wave & 3) * 16;   // row group (stage1 h-tile & stage2)
    const int cw = (wave >> 2) * 16;  // stage1 h-col group
    const int ch = wave >> 2;         // stage2 column half

    // ---- stage x tile once (64 x E, f32 -> f16, A layout) ----
#pragma unroll
    for (int j = 0; j < E / 16; ++j) {
        int idx = j * 256 + tid;
        int r = idx / (E / 4), c = (idx % (E / 4)) * 4;
        v4f v = *(const v4f*)(X + (size_t)(bm + r) * E + c);
        v4h h = {(_Float16)v[0], (_Float16)v[1], (_Float16)v[2],
                 (_Float16)v[3]};
        *(v4h*)&xs[r][c] = h;
    }
    __syncthreads();

    v8f acc[E / 32];
#pragma unroll
    for (int i = 0; i < E / 32; ++i)
        acc[i] = (v8f){0.f, 0.f, 0.f, 0.f, 0.f, 0.f, 0.f, 0.f};

    for (int hc = 0; hc < 4 * E; hc += 32) {
        // ---- stage W1[:, hc..hc+32) transposed -> wbuf[c*XP + k] ----
#pragma unroll
        for (int j = 0; j < E / 32; ++j) {
            int c = tid & 31;
            int k = ((tid >> 5) + j * 8) * 4;
            v4h h = {(_Float16)W1[(size_t)(k + 0) * (4 * E) + hc + c],
                     (_Float16)W1[(size_t)(k + 1) * (4 * E) + hc + c],
                     (_Float16)W1[(size_t)(k + 2) * (4 * E) + hc + c],
                     (_Float16)W1[(size_t)(k + 3) * (4 * E) + hc + c]};
            *(v4h*)&wbuf[c * XP + k] = h;
        }
        __syncthreads();

        // ---- stage 1: h(64x32) = relu(x @ W1chunk + b1chunk) ----
        v8f hacc = (v8f){0.f, 0.f, 0.f, 0.f, 0.f, 0.f, 0.f, 0.f};
#pragma unroll
        for (int kc = 0; kc < E / 32; ++kc) {
            int kb = kc * 32;
            v8h x0 = *(const v8h*)&xs[rw + l][kb + hi * 8];
            v8h x1 = *(const v8h*)&xs[rw + l][kb + 16 + hi * 8];
            v8h z0 = *(const v8h*)&wbuf[(cw + l) * XP + kb + hi * 8];
            v8h z1 = *(const v8h*)&wbuf[(cw + l) * XP + kb + 16 + hi * 8];
            hacc = WMMA(CAT16(x0, x1), CAT16(z0, z1), hacc);
        }
        const float bb = b1[hc + cw + l];
#pragma unroll
        for (int r = 0; r < 8; ++r)
            hs[rw + hi * 8 + r][cw + l] = (_Float16)fmaxf(hacc[r] + bb, 0.f);
        __syncthreads();

        // ---- stage W2[hc..hc+32, :] transposed -> wbuf[c*HP + k] ----
#pragma unroll
        for (int j = 0; j < E / 32; ++j) {
            int idx = j * 256 + tid;
            int c  = idx & (E - 1);
            int k  = (idx / E) * 4;
            v4h h = {(_Float16)W2[(size_t)(hc + k + 0) * E + c],
                     (_Float16)W2[(size_t)(hc + k + 1) * E + c],
                     (_Float16)W2[(size_t)(hc + k + 2) * E + c],
                     (_Float16)W2[(size_t)(hc + k + 3) * E + c]};
            *(v4h*)&wbuf[c * HP + k] = h;
        }
        __syncthreads();

        // ---- stage 2: acc += h @ W2chunk ----
        {
            v8h a0 = *(const v8h*)&hs[rw + l][hi * 8];
            v8h a1 = *(const v8h*)&hs[rw + l][16 + hi * 8];
            v16h a = CAT16(a0, a1);
#pragma unroll
            for (int ct = 0; ct < E / 32; ++ct) {
                int cb = ch * (E / 2) + ct * 16;
                v8h z0 = *(const v8h*)&wbuf[(cb + l) * HP + hi * 8];
                v8h z1 = *(const v8h*)&wbuf[(cb + l) * HP + 16 + hi * 8];
                acc[ct] = WMMA(a, CAT16(z0, z1), acc[ct]);
            }
        }
        __syncthreads();
    }

    // ---- epilogue: add b2, store f32 ----
#pragma unroll
    for (int ct = 0; ct < E / 32; ++ct) {
        int col = ch * (E / 2) + ct * 16 + l;
        const float bv = b2[col];
#pragma unroll
        for (int r = 0; r < 8; ++r) {
            int row = bm + rw + hi * 8 + r;
            Out[(size_t)row * E + col] = acc[ct][r] + bv;
        }
    }
}

// ------------------------ fused residual + LayerNorm -----------------------
__global__ void __launch_bounds__(256)
ln_kernel(const float* __restrict__ a, const float* __restrict__ b,
          const float* __restrict__ g, const float* __restrict__ be,
          float* __restrict__ out, int E)
{
    __shared__ float red[256];
    __shared__ float s_mu, s_rstd;
    const int row = blockIdx.x;
    const int t   = threadIdx.x;

    float v = 0.f;
    if (t < E) {
        v = a[(size_t)row * E + t];
        if (b) v += b[(size_t)row * E + t];
    }
    red[t] = (t < E) ? v : 0.f;
    __syncthreads();
    for (int s = 128; s > 0; s >>= 1) {
        if (t < s) red[t] += red[t + s];
        __syncthreads();
    }
    if (t == 0) s_mu = red[0] / (float)E;
    __syncthreads();
    const float mu = s_mu;
    const float d  = (t < E) ? (v - mu) : 0.f;
    red[t] = d * d;
    __syncthreads();
    for (int s = 128; s > 0; s >>= 1) {
        if (t < s) red[t] += red[t + s];
        __syncthreads();
    }
    if (t == 0) s_rstd = rsqrtf(red[0] / (float)E + 1e-5f);
    __syncthreads();
    if (t < E) out[(size_t)row * E + t] = d * s_rstd * g[t] + be[t];
}

// --------------------------- spatial attention -----------------------------
// maskMode: 0 none; 1 adj_sr.T (mask[q][k] = adj_sr[k*Qn+q]); 2 adj_r[q*Kn+k]
__global__ void __launch_bounds__(256)
spatial_attn(const float* __restrict__ qq, const float* __restrict__ kk,
             const float* __restrict__ vv, const float* __restrict__ mask,
             int maskMode, float* __restrict__ out,
             int Qn, int Kn, int Tdim, float scale)
{
    __shared__ float qsh[16];
    __shared__ float psum[256];
    __shared__ float accs[16][256];

    const int q   = blockIdx.x;
    const int t   = blockIdx.y;
    const int tid = threadIdx.x;

    for (int h = 0; h < H_; ++h) {
        if (tid < 16)
            qsh[tid] = qq[(((size_t)q * Tdim + t) * H_ + h) * 16 + tid];
        __syncthreads();
        float qr[16];
#pragma unroll
        for (int dd = 0; dd < 16; ++dd) qr[dd] = qsh[dd];

        float myacc[16];
#pragma unroll
        for (int dd = 0; dd < 16; ++dd) myacc[dd] = 0.f;
        float mysum = 0.f;

        for (int k = tid; k < Kn; k += 256) {
            const float* kp = kk + (((size_t)k * Tdim + t) * H_ + h) * 16;
            float e = 0.f;
#pragma unroll
            for (int dd = 0; dd < 16; ++dd) e += qr[dd] * kp[dd];
            if (maskMode == 1) {
                if (mask[(size_t)k * Qn + q] == 0.f) e = NEGV;
            } else if (maskMode == 2) {
                if (mask[(size_t)q * Kn + k] == 0.f) e = NEGV;
            }
            e = fminf(fmaxf(e * scale, -5.f), 5.f);  // clip; NEG -> -5
            const float p = expf(e);                 // no max pass needed
            mysum += p;
            const float* vp = vv + (((size_t)k * Tdim + t) * H_ + h) * 16;
#pragma unroll
            for (int dd = 0; dd < 16; ++dd) myacc[dd] += p * vp[dd];
        }

        psum[tid] = mysum;
#pragma unroll
        for (int dd = 0; dd < 16; ++dd) accs[dd][tid] = myacc[dd];
        __syncthreads();
        for (int s = 128; s > 0; s >>= 1) {
            if (tid < s) {
                psum[tid] += psum[tid + s];
#pragma unroll
                for (int dd = 0; dd < 16; ++dd)
                    accs[dd][tid] += accs[dd][tid + s];
            }
            __syncthreads();
        }
        if (tid < 16)
            out[(((size_t)q * Tdim + t) * H_ + h) * 16 + tid] =
                accs[tid][0] / psum[0];
        __syncthreads();
    }
}

// --------------------------- temporal attention ----------------------------
__global__ void __launch_bounds__(128)
temporal_attn(const float* __restrict__ qq, const float* __restrict__ kk,
              const float* __restrict__ vv, float* __restrict__ out,
              int Tq, int Tk, int d, int causal, float scale)
{
    __shared__ float qsh[18 * 32];
    __shared__ float ksh[18 * 32];
    __shared__ float vsh[18 * 32];
    __shared__ float esh[18 * 18];
    __shared__ float rsum[18];

    const int n   = blockIdx.x;
    const int h   = blockIdx.y;
    const int tid = threadIdx.x;

    for (int i = tid; i < Tq * d; i += 128) {
        int tq = i / d, dd = i % d;
        qsh[i] = qq[(((size_t)n * Tq + tq) * H_ + h) * d + dd];
    }
    for (int i = tid; i < Tk * d; i += 128) {
        int tk = i / d, dd = i % d;
        ksh[i] = kk[(((size_t)n * Tk + tk) * H_ + h) * d + dd];
        vsh[i] = vv[(((size_t)n * Tk + tk) * H_ + h) * d + dd];
    }
    __syncthreads();

    for (int i = tid; i < Tq * Tk; i += 128) {
        int tq = i / Tk, tk = i % Tk;
        float e = 0.f;
        for (int dd = 0; dd < d; ++dd) e += qsh[tq * d + dd] * ksh[tk * d + dd];
        if (causal && tk > tq) e = NEGV;
        e = fminf(fmaxf(e * scale, -5.f), 5.f);
        esh[tq * 18 + tk] = expf(e);
    }
    __syncthreads();
    for (int i = tid; i < Tq; i += 128) {
        float s = 0.f;
        for (int tk = 0; tk < Tk; ++tk) s += esh[i * 18 + tk];
        rsum[i] = s;
    }
    __syncthreads();
    for (int i = tid; i < Tq * d; i += 128) {
        int tq = i / d, dd = i % d;
        float o = 0.f;
        for (int tk = 0; tk < Tk; ++tk) o += esh[tq * 18 + tk] * vsh[tk * d + dd];
        out[(((size_t)n * Tq + tq) * H_ + h) * d + dd] = o / rsum[tq];
    }
}

// ------------------------------ elementwise --------------------------------
__global__ void add_ds_kernel(const float* __restrict__ src,
                              const float* __restrict__ ds,
                              float* __restrict__ out, int total)
{
    int i = blockIdx.x * blockDim.x + threadIdx.x;
    if (i >= total) return;
    int n = i / (T_ * E_);
    int c = i % E_;
    out[i] = src[i] + ds[(size_t)n * E_ + c];
}

__global__ void cat2_kernel(const float* __restrict__ a,
                            const float* __restrict__ b,
                            float* __restrict__ out, int M, int Eh)
{
    int i = blockIdx.x * blockDim.x + threadIdx.x;
    int total = M * 2 * Eh;
    if (i >= total) return;
    int row = i / (2 * Eh);
    int c   = i % (2 * Eh);
    out[i] = (c < Eh) ? a[(size_t)row * Eh + c]
                      : (b ? b[(size_t)row * Eh + (c - Eh)] : 0.f);
}

__global__ void make_tgt1_kernel(const float* __restrict__ senc,
                                 float* __restrict__ out)
{
    int i = blockIdx.x * blockDim.x + threadIdx.x;
    int total = N_ * 6 * E2_;
    if (i >= total) return;
    int row = i / E2_;
    int n = row / 6, tq = row % 6;
    int c = i % E2_;
    out[i] = senc[((size_t)n * T_ + 12 + tq) * E_ + (c % E_)];
}

__global__ void abs_tile_kernel(const float* __restrict__ senc,
                                const float* __restrict__ x1,
                                float* __restrict__ cout)
{
    int i = blockIdx.x * blockDim.x + threadIdx.x;
    int total = N_ * T_ * E_;
    if (i >= total) return;
    int row = i / E_;
    int n = row / T_, t = row % T_;
    int c = i % E_;
    cout[i] = fabsf(senc[i] - x1[((size_t)n * 6 + (t % 6)) * E_ + c]);
}

// ------------------------------ host helpers -------------------------------
struct AttnP { const float *Wv, *Wk, *Wq, *Wo, *bo; };
struct TTP   { AttnP att; const float *n1g, *n1b, *n2g, *n2b, *W1, *b1, *W2, *b2; };
struct DecP  { TTP tt_self, tt_mu; const float *n1g, *n1b, *n2g, *n2b; };

static inline void gemm(hipStream_t s, const float* A, const float* B,
                        const float* bias, float* C, int M, int Nc, int K,
                        int act)
{
    dim3 g((M + BM - 1) / BM, (Nc + BN - 1) / BN);
    gemm_f16wmma<<<g, 256, 0, s>>>(A, B, bias, C, M, Nc, K, act);
}

static inline void ffn(hipStream_t s, const float* x, const float* W1,
                       const float* b1, const float* W2, const float* b2,
                       float* out, int M, int e)
{
    if (e == 128)
        ffn_fused<128><<<M / 64, 256, 0, s>>>(x, W1, b1, W2, b2, out);
    else
        ffn_fused<256><<<M / 64, 256, 0, s>>>(x, W1, b1, W2, b2, out);
}

static inline void lnrun(hipStream_t s, const float* a, const float* b,
                         const float* g, const float* be, float* out, int M,
                         int E)
{
    ln_kernel<<<M, 256, 0, s>>>(a, b, g, be, out, E);
}

// _s_attn (E=128, d=16)
static void run_sattn(hipStream_t s, const AttnP& p, const float* v,
                      const float* k, const float* q, int Kn, int Qn,
                      const float* mask, int maskMode, float* out, float* pv,
                      float* pk, float* pq, float* pattn)
{
    gemm(s, v, p.Wv, nullptr, pv, Kn * T_ * H_, 16, 16, 0);
    gemm(s, k, p.Wk, nullptr, pk, Kn * T_ * H_, 16, 16, 0);
    gemm(s, q, p.Wq, nullptr, pq, Qn * T_ * H_, 16, 16, 0);
    spatial_attn<<<dim3(Qn, T_), 256, 0, s>>>(pq, pk, pv, mask, maskMode,
                                              pattn, Qn, Kn, T_,
                                              1.f / sqrtf((float)E_));
    gemm(s, pattn, p.Wo, p.bo, out, Qn * T_, E_, E_, 0);
}

// _tt (temporal transformer block), e = E2 in all uses
static void run_tt(hipStream_t s, const TTP& p, const float* v, const float* k,
                   const float* q, int Nb, int Tk, int Tq, bool causal, int e,
                   float* out, float* pv, float* pk, float* pq, float* pattn,
                   float* s1, float* s2)
{
    const int d = e / H_;
    gemm(s, v, p.att.Wv, nullptr, pv, Nb * Tk * H_, d, d, 0);
    gemm(s, k, p.att.Wk, nullptr, pk, Nb * Tk * H_, d, d, 0);
    gemm(s, q, p.att.Wq, nullptr, pq, Nb * Tq * H_, d, d, 0);
    temporal_attn<<<dim3(Nb, H_), 128, 0, s>>>(pq, pk, pv, pattn, Tq, Tk, d,
                                               causal ? 1 : 0,
                                               1.f / sqrtf((float)e));
    gemm(s, pattn, p.att.Wo, p.att.bo, s1, Nb * Tq, e, e, 0);   // a
    lnrun(s, s1, q, p.n1g, p.n1b, s2, Nb * Tq, e);              // x = ln(a+q)
    ffn(s, s2, p.W1, p.b1, p.W2, p.b2, s1, Nb * Tq, e);         // fused FFN
    lnrun(s, s1, s2, p.n2g, p.n2b, out, Nb * Tq, e);            // ln(ffn+x)
}

static void run_decoder(hipStream_t s, const DecP& p, const float* xmem,
                        const float* tgt1, float* out, float* pv, float* pk,
                        float* pq, float* pattn, float* s1, float* s2,
                        float* dA, float* dB)
{
    run_tt(s, p.tt_self, tgt1, tgt1, tgt1, N_, 6, 6, true, E2_, dA, pv, pk,
           pq, pattn, s1, s2);
    lnrun(s, dA, tgt1, p.n1g, p.n1b, dB, N_ * 6, E2_);  // x1 = ln(x+query)
    run_tt(s, p.tt_mu, xmem, xmem, dB, N_, T_, 6, false, E2_, dA, pv, pk, pq,
           pattn, s1, s2);
    lnrun(s, dA, dB, p.n2g, p.n2b, out, N_ * 6, E2_);
}

// ------------------------------ entry point --------------------------------
extern "C" void kernel_launch(void* const* d_in, const int* in_sizes, int n_in,
                              void* d_out, int out_size, void* d_ws,
                              size_t ws_size, hipStream_t stream)
{
    (void)in_sizes; (void)n_in; (void)out_size; (void)ws_size;

    // --- parse inputs: setup_inputs() dict insertion order, recursively ---
    int idx = 0;
    auto nf = [&]() { return (const float*)d_in[idx++]; };
    auto rdAttn = [&]() {
        AttnP a; a.Wv = nf(); a.Wk = nf(); a.Wq = nf(); a.Wo = nf(); a.bo = nf();
        return a;
    };
    auto rdTT = [&]() {
        TTP t; t.att = rdAttn();
        t.n1g = nf(); t.n1b = nf(); t.n2g = nf(); t.n2b = nf();
        t.W1 = nf(); t.b1 = nf(); t.W2 = nf(); t.b2 = nf();
        return t;
    };
    auto rdDec = [&]() {
        DecP d; d.tt_self = rdTT(); d.tt_mu = rdTT();
        d.n1g = nf(); d.n1b = nf(); d.n2g = nf(); d.n2b = nf();
        return d;
    };

    const float* src = nf();
    idx++;  // 't' scalar, unused
    // params.st
    const float* D_S  = nf();
    const float* embW = nf();
    const float* embb = nf();
    const float* I_   = nf();
    AttnP att_sr = rdAttn();
    AttnP att_rr = rdAttn();
    AttnP att_rs = rdAttn();
    const float* stn1g = nf(); const float* stn1b = nf();
    const float* stn2g = nf(); const float* stn2b = nf();
    const float* stW1 = nf();  const float* stb1 = nf();
    const float* stW2 = nf();  const float* stb2 = nf();
    const float* fsW = nf();   const float* fsb = nf();
    // params (rest)
    const float* t2n1g = nf(); const float* t2n1b = nf();
    const float* t2n2g = nf(); const float* t2n2b = nf();
    TTP  tt_full = rdTT();
    DecP dec1 = rdDec();
    DecP dec2 = rdDec();
    const float* fcnW = nf(); const float* fcnb = nf();
    const float* lnW  = nf(); const float* lnb  = nf();
    const float* adj_r  = nf();
    const float* adj_sr = nf();

    // --- workspace carve-up ---
    float* w = (float*)d_ws;
    size_t off = 0;
    auto alloc = [&](size_t n) { float* p = w + off; off += n; return p; };
    const size_t NT = (size_t)N_ * T_;           // 36864
    const size_t N6 = (size_t)N_ * 6;            // 12288

    float* dsb   = alloc((size_t)N_ * E_);
    float* qbuf  = alloc(NT * E_);
    float* senc  = alloc(NT * E_);
    float* hrA   = alloc((size_t)R_ * T_ * E_);
    float* hrB   = alloc((size_t)R_ * T_ * E_);
    float* pv    = alloc(NT * E2_);
    float* pk    = alloc(NT * E2_);
    float* pq    = alloc(NT * E2_);
    float* pattn = alloc(NT * E2_);
    float* s1    = alloc(NT * E2_);
    float* s2    = alloc(NT * E2_);
    float* src1  = alloc(NT * E2_);
    float* ttout = alloc(NT * E2_);
    float* xB    = alloc(NT * E2_);
    float* tgt1  = alloc(N6 * E2_);
    float* dA    = alloc(N6 * E2_);
    float* dB    = alloc(N6 * E2_);
    float* dC    = alloc(N6 * E2_);
    float* x1buf = alloc(N6 * E_);
    float* x2buf = alloc(N6 * E_);

    // --- outputs (flat, in return order) ---
    float* out_x  = (float*)d_out;               // (N,T,E2)
    float* out_c  = out_x + NT * E2_;            // (N,T,E)
    float* out_o3 = out_c + NT * E_;             // (N,6,2)
    float* out_o4 = out_o3 + N6 * 2;             // (N,6,2)

    const int TPB = 256;
    auto grid1 = [&](size_t n) { return dim3((unsigned)((n + TPB - 1) / TPB)); };

    // ================= Stage 1: spatial transformer (E=128) ================
    gemm(stream, D_S, embW, embb, dsb, N_, E_, R_, 0);            // DS
    add_ds_kernel<<<grid1(NT * E_), TPB, 0, stream>>>(src, dsb, qbuf,
                                                      (int)(NT * E_));
    run_sattn(stream, att_sr, qbuf, qbuf, I_, N_, R_, adj_sr, 1, hrA, pv, pk,
              pq, pattn);
    run_sattn(stream, att_rr, hrA, hrA, hrA, R_, R_, adj_r, 2, hrB, pv, pk,
              pq, pattn);
    run_sattn(stream, att_rs, hrB, hrB, qbuf, R_, N_, nullptr, 0, s1, pv, pk,
              pq, pattn);
    lnrun(stream, s1, qbuf, stn1g, stn1b, s2, (int)NT, E_);       // x
    ffn(stream, s2, stW1, stb1, stW2, stb2, s1, (int)NT, E_);     // fused FFN
    lnrun(stream, s1, s2, stn2g, stn2b, pq, (int)NT, E_);         // US
    gemm(stream, pq, fsW, fsb, s1, (int)NT, E_, E_, 0);           // st_out
    lnrun(stream, s1, src, t2n1g, t2n1b, senc, (int)NT, E_);      // senc

    // =================== Stage 2: two encode+decode passes ==================
    auto encode_pass = [&](const float* catB, float* xEncOut) {
        cat2_kernel<<<grid1(NT * E2_), TPB, 0, stream>>>(senc, catB, src1,
                                                         (int)NT, E_);
        run_tt(stream, tt_full, src1, src1, src1, N_, T_, T_, false, E2_,
               ttout, pv, pk, pq, pattn, s1, s2);
        lnrun(stream, ttout, src1, t2n2g, t2n2b, xEncOut, (int)NT, E2_);
    };

    // pass 1 (src1 = [senc, 0])
    encode_pass(nullptr, xB);
    make_tgt1_kernel<<<grid1(N6 * E2_), TPB, 0, stream>>>(senc, tgt1);
    run_decoder(stream, dec1, xB, tgt1, dC, pv, pk, pq, pattn, s1, s2, dA, dB);
    gemm(stream, dC, fcnW, fcnb, x1buf, (int)N6, E_, E2_, 2);     // sigmoid
    abs_tile_kernel<<<grid1(NT * E_), TPB, 0, stream>>>(senc, x1buf, out_c);

    // pass 2 (src2 = [senc, c]); final x written straight to d_out
    encode_pass(out_c, out_x);
    run_decoder(stream, dec2, out_x, tgt1, dC, pv, pk, pq, pattn, s1, s2,
                dA, dB);
    gemm(stream, dC, fcnW, fcnb, x2buf, (int)N6, E_, E2_, 2);     // sigmoid

    // final linear heads (N*6 x 2)
    gemm(stream, x1buf, lnW, lnb, out_o3, (int)N6, 2, E_, 0);
    gemm(stream, x2buf, lnW, lnb, out_o4, (int)N6, 2, E_, 0);
}
